// EncoderLayer_46703474377479
// MI455X (gfx1250) — matrix operand order repaired and checked by reading, not compiled, not run on hardware
//
#include <hip/hip_runtime.h>

// ---------------------------------------------------------------------------
// CDNA5 (gfx1250) transformer encoder layer, bf16 WMMA path,
// double-buffered async global->LDS staging (ASYNCcnt DMA).
// B=4, S=2048, D_MODEL=1024, H=16, Dk=64, D_FF=4096. wave32.
// ---------------------------------------------------------------------------

typedef __bf16 bf16_t;
typedef __attribute__((ext_vector_type(16))) __bf16 v16bf;
typedef __attribute__((ext_vector_type(8)))  float  v8f;

#define S_LEN   2048
#define D_MODEL 1024
#define N_HEADS 16
#define D_K     64
#define D_FF    4096
#define BATCH   4
#define ROWS    (BATCH * S_LEN)   // 8192

// ---- bf16 helpers (manual RNE, storage-safe) ------------------------------
__device__ __forceinline__ bf16_t f2bf(float f) {
    unsigned u = __builtin_bit_cast(unsigned, f);
    unsigned r = u + 0x7FFFu + ((u >> 16) & 1u);
    unsigned short hs = (unsigned short)(r >> 16);
    bf16_t out;
    __builtin_memcpy(&out, &hs, 2);
    return out;
}

__device__ __forceinline__ v8f v8f_zero() {
    v8f z;
#pragma unroll
    for (int i = 0; i < 8; ++i) z[i] = 0.0f;
    return z;
}

__device__ __forceinline__ v8f wmma_bf16(v16bf a, v16bf b, v8f c) {
    return __builtin_amdgcn_wmma_f32_16x16x32_bf16(
        false, a, false, b, (short)0, c, false, false);
}

// A-fragment (16x32 bf16): element e<8 <-> k = 8*hlf+e ; e>=8 <-> k = 16+8*hlf+(e-8)
__device__ __forceinline__ v16bf load_afrag(const bf16_t* rowp, int hlf) {
    v16bf a;
#pragma unroll
    for (int j = 0; j < 8; ++j) {
        a[j]     = rowp[8 * hlf + j];
        a[8 + j] = rowp[16 + 8 * hlf + j];
    }
    return a;
}

// B-fragment (32x16 bf16): element e<8 <-> k = 8*hlf+e ; e>=8 <-> k = 16+8*hlf+(e-8)
__device__ __forceinline__ v16bf load_bfrag(const bf16_t* colp, int hlf) {
    v16bf b;
#pragma unroll
    for (int j = 0; j < 8; ++j) {
        b[j]     = colp[8 * hlf + j];
        b[8 + j] = colp[16 + 8 * hlf + j];
    }
    return b;
}

// ---- CDNA5 async global->LDS (ASYNCcnt path) ------------------------------
// Flat LDS addresses carry the LDS byte address in the low 32 bits,
// so truncation yields the DS address.
__device__ __forceinline__ void async_copy_b128(void* lds_dst, const void* gsrc) {
    unsigned l = (unsigned)(uintptr_t)lds_dst;
    asm volatile("global_load_async_to_lds_b128 %0, %1, off"
                 :: "v"(l), "v"(gsrc) : "memory");
}
__device__ __forceinline__ void async_wait0() {
    asm volatile("s_wait_asynccnt 0" ::: "memory");
}

// ---------------------------------------------------------------------------
// Tiled transpose + f32->bf16: src[K][N] f32  ->  dst[N][K] bf16
// ---------------------------------------------------------------------------
__global__ __launch_bounds__(256) void cvt_transpose_kernel(const float* __restrict__ src,
                                                            bf16_t* __restrict__ dst,
                                                            int K, int N) {
    __shared__ float t[32][33];
    const int k0 = blockIdx.y * 32, n0 = blockIdx.x * 32;
    const int tx = threadIdx.x & 31, ty = threadIdx.x >> 5;   // ty: 0..7
#pragma unroll
    for (int r = ty; r < 32; r += 8)
        t[r][tx] = src[(size_t)(k0 + r) * N + n0 + tx];
    __syncthreads();
#pragma unroll
    for (int r = ty; r < 32; r += 8)
        dst[(size_t)(n0 + r) * K + k0 + tx] = f2bf(t[tx][r]);
}

// ---------------------------------------------------------------------------
// LayerNorm over 1024 cols, one block per row, bf16 output
// ---------------------------------------------------------------------------
__global__ __launch_bounds__(256) void ln_bf16_kernel(const float* __restrict__ x,
                                                      const float* __restrict__ g,
                                                      const float* __restrict__ b,
                                                      bf16_t* __restrict__ out) {
    const int tid = threadIdx.x;
    const int lane = tid & 31, wid = tid >> 5;
    const size_t row = blockIdx.x;
    const float* xr = x + row * D_MODEL;

    float vals[4];
    float lsum = 0.f, lsq = 0.f;
#pragma unroll
    for (int i = 0; i < 4; ++i) {
        float t = xr[tid + 256 * i];
        vals[i] = t;
        lsum += t;
        lsq += t * t;
    }
#pragma unroll
    for (int off = 1; off < 32; off <<= 1) {
        lsum += __shfl_xor(lsum, off, 32);
        lsq  += __shfl_xor(lsq,  off, 32);
    }
    __shared__ float ws1[8], ws2[8];
    if (lane == 0) { ws1[wid] = lsum; ws2[wid] = lsq; }
    __syncthreads();
    if (tid == 0) {
        float s = 0.f, q = 0.f;
#pragma unroll
        for (int w = 0; w < 8; ++w) { s += ws1[w]; q += ws2[w]; }
        ws1[0] = s * (1.0f / D_MODEL);
        ws2[0] = q * (1.0f / D_MODEL);
    }
    __syncthreads();
    const float mu   = ws1[0];
    const float var  = ws2[0] - mu * mu;
    const float rstd = rsqrtf(var + 1e-5f);
#pragma unroll
    for (int i = 0; i < 4; ++i) {
        int c = tid + 256 * i;
        out[row * D_MODEL + c] = f2bf((vals[i] - mu) * rstd * g[c] + b[c]);
    }
}

// ---------------------------------------------------------------------------
// bf16 GEMM with transposed-B weights: C[M,N] = act(A[M,K] @ BT[N,K]^T + bias)
// Block tile 128x128, BK=32; 8 waves, each 32x64 (2x4 WMMA accumulators).
// Double-buffered async-LDS staging: DMA of tile i+1 overlaps WMMA of tile i.
// Epilogue modes: f32 out (+residual), bf16 out, or V-transposed bf16 out.
// All dims are multiples of the tiles.
// ---------------------------------------------------------------------------
__global__ __launch_bounds__(256) void gemm_bf16_kernel(
    const bf16_t* __restrict__ A, int lda,
    const bf16_t* __restrict__ BT, int ldb,
    const float* __restrict__ bias,
    const float* __restrict__ residual, int ldr,
    float* __restrict__ Cf, bf16_t* __restrict__ Cb, bf16_t* __restrict__ vt,
    int ldc, int K, int act) {
    __shared__ __align__(16) bf16_t As[2][128 * 48];   // [row][k] pitch 48 (96B)
    __shared__ __align__(16) bf16_t Bs[2][128 * 48];   // [col][k] pitch 48 (96B)

    const int tid  = threadIdx.x;
    const int lane = tid & 31;
    const int wid  = tid >> 5;
    const int wm   = wid >> 1;        // 0..3
    const int wn   = wid & 1;         // 0..1
    const int hlf  = lane >> 4;
    const int l16  = lane & 15;
    const int bm   = blockIdx.y * 128;
    const int bn   = blockIdx.x * 128;

    // per-thread staging chunk coordinates (2 x 16B per matrix)
    const int r0 = tid >> 2,          kc0 = (tid & 3) * 8;        // chunks 0..255
    const int r1 = (tid + 256) >> 2,  kc1 = ((tid + 256) & 3) * 8;

    auto issue_tiles = [&](int buf, int k0) {
        async_copy_b128(&As[buf][r0 * 48 + kc0], &A[(size_t)(bm + r0) * lda + k0 + kc0]);
        async_copy_b128(&Bs[buf][r0 * 48 + kc0], &BT[(size_t)(bn + r0) * ldb + k0 + kc0]);
        async_copy_b128(&As[buf][r1 * 48 + kc1], &A[(size_t)(bm + r1) * lda + k0 + kc1]);
        async_copy_b128(&Bs[buf][r1 * 48 + kc1], &BT[(size_t)(bn + r1) * ldb + k0 + kc1]);
    };

    v8f acc[2][4];
#pragma unroll
    for (int i = 0; i < 2; ++i)
#pragma unroll
        for (int j = 0; j < 4; ++j) acc[i][j] = v8f_zero();

    issue_tiles(0, 0);   // prologue DMA

    for (int k0 = 0; k0 < K; k0 += 32) {
        const int cur = (k0 >> 5) & 1;
        async_wait0();       // my writes of buf[cur] have landed
        __syncthreads();     // everyone's have; buf[cur^1] readers are done
        if (k0 + 32 < K) issue_tiles(cur ^ 1, k0 + 32);   // overlap with WMMA below

        v16bf af[2], bfr[4];
#pragma unroll
        for (int mt = 0; mt < 2; ++mt)
            af[mt] = load_afrag(&As[cur][(wm * 32 + mt * 16 + l16) * 48], hlf);
#pragma unroll
        for (int nt = 0; nt < 4; ++nt)
            bfr[nt] = load_bfrag(&Bs[cur][(wn * 64 + nt * 16 + l16) * 48], hlf);
#pragma unroll
        for (int mt = 0; mt < 2; ++mt)
#pragma unroll
            for (int nt = 0; nt < 4; ++nt)
                acc[mt][nt] = wmma_bf16(af[mt], bfr[nt], acc[mt][nt]);
    }

    // epilogue: bias -> activation -> residual -> store
#pragma unroll
    for (int mt = 0; mt < 2; ++mt) {
#pragma unroll
        for (int nt = 0; nt < 4; ++nt) {
            const int col = bn + wn * 64 + nt * 16 + l16;
            const float bv = bias ? bias[col] : 0.0f;
#pragma unroll
            for (int v = 0; v < 8; ++v) {
                const int row = bm + wm * 32 + mt * 16 + v + 8 * hlf;
                float val = acc[mt][nt][v] + bv;
                if (act == 1)  // exact erf GELU
                    val = 0.5f * val * (1.0f + erff(val * 0.70710678118654752f));
                if (residual) val += residual[(size_t)row * ldr + col];
                if (Cf) Cf[(size_t)row * ldc + col] = val;
                if (Cb) Cb[(size_t)row * ldc + col] = f2bf(val);
                if (vt) {   // V stored transposed: [b][h][d][s]
                    const int bb = row >> 11, s = row & 2047;
                    const int hh = col >> 6,  d = col & 63;
                    vt[(((size_t)bb * N_HEADS + hh) * D_K + d) * S_LEN + s] = f2bf(val);
                }
            }
        }
    }
}

// ---------------------------------------------------------------------------
// Flash attention: grid (S/128, B*H); 8 waves, 16 query rows per wave.
// qk : bf16 [ROWS, 2048] = [Q | K], each [ROWS, 16*64]
// vt : bf16 [B][H][64][S]  (d-major V)
// out: bf16 [ROWS, 1024]  (heads concatenated)
// Double-buffered async-LDS staging of K/V blocks.
// ---------------------------------------------------------------------------
__global__ __launch_bounds__(256) void attn_kernel(const bf16_t* __restrict__ qk,
                                                   const bf16_t* __restrict__ vt,
                                                   const int* __restrict__ mask,
                                                   bf16_t* __restrict__ outp) {
    const int tid  = threadIdx.x;
    const int lane = tid & 31;
    const int wid  = tid >> 5;
    const int hlf  = lane >> 4;
    const int l16  = lane & 15;
    const int bh   = blockIdx.y;          // 0..63
    const int b    = bh >> 4;
    const int h    = bh & (N_HEADS - 1);
    const int q0   = blockIdx.x * 128 + wid * 16;

    const bf16_t* Qp = qk + ((size_t)b * S_LEN) * 2048 + (size_t)h * D_K;
    const bf16_t* Kp = Qp + 1024;
    const bf16_t* Vp = vt + (size_t)bh * D_K * S_LEN;   // row d: 2048 s-values

    __shared__ __align__(16) bf16_t Kb[2][32 * 72];    // [key][d]   pitch 72 (144B)
    __shared__ __align__(16) bf16_t Vbt[2][64 * 48];   // [d][key]   pitch 48 (96B)
    __shared__ __align__(16) bf16_t Ps[128 * 40];      // [qrow][k]  pitch 40 (80B)

    const int kr = tid >> 3, kcc = (tid & 7) * 8;   // K: 256 chunks, 1/thread
    const int vr = tid >> 2, vcc = (tid & 3) * 8;   // V: 256 chunks, 1/thread

    auto issue_kv = [&](int buf, int kb) {
        async_copy_b128(&Kb[buf][kr * 72 + kcc], &Kp[(size_t)(kb + kr) * 2048 + kcc]);
        async_copy_b128(&Vbt[buf][vr * 48 + vcc], &Vp[(size_t)vr * S_LEN + kb + vcc]);
    };

    // Q A-fragments (constant over the key loop), d-steps 0 and 1
    v16bf aq[2];
#pragma unroll
    for (int ds = 0; ds < 2; ++ds)
        aq[ds] = load_afrag(&Qp[(size_t)(q0 + l16) * 2048 + ds * 32], hlf);

    v8f acc[4];
#pragma unroll
    for (int dt = 0; dt < 4; ++dt) acc[dt] = v8f_zero();
    float mrow[8], lrow[8];
#pragma unroll
    for (int v = 0; v < 8; ++v) { mrow[v] = -1e30f; lrow[v] = 0.0f; }

    issue_kv(0, 0);   // prologue DMA

    for (int kb = 0; kb < S_LEN; kb += 32) {
        const int cur = (kb >> 5) & 1;
        async_wait0();
        __syncthreads();   // buf[cur] ready everywhere; buf[cur^1] readers done
        if (kb + 32 < S_LEN) issue_kv(cur ^ 1, kb + 32);   // overlap with compute

        // S tile 16x32 = two 16x16 WMMA n-tiles, inner dim 64 (2 d-steps)
        v8f s_t[2];
#pragma unroll
        for (int nt = 0; nt < 2; ++nt) {
            v8f c = v8f_zero();
#pragma unroll
            for (int ds = 0; ds < 2; ++ds)
                c = wmma_bf16(aq[ds],
                              load_bfrag(&Kb[cur][(nt * 16 + l16) * 72 + ds * 32], hlf),
                              c);
            s_t[nt] = c;
        }

        const bool mz0 = (mask[(size_t)b * S_LEN + kb + l16] == 0);
        const bool mz1 = (mask[(size_t)b * S_LEN + kb + 16 + l16] == 0);

        // online softmax per row (rows live in 16-lane halves)
#pragma unroll
        for (int v = 0; v < 8; ++v) {
            float s0 = s_t[0][v] * 0.125f;       // 1/sqrt(64)
            float s1 = s_t[1][v] * 0.125f;
            if (mz0) s0 = -1e9f;
            if (mz1) s1 = -1e9f;
            float r = fmaxf(s0, s1);
#pragma unroll
            for (int off = 1; off < 16; off <<= 1) r = fmaxf(r, __shfl_xor(r, off, 32));
            const float mn = fmaxf(mrow[v], r);
            const float sc = __expf(mrow[v] - mn);
            const float p0 = __expf(s0 - mn);
            const float p1 = __expf(s1 - mn);
            float rs = p0 + p1;
#pragma unroll
            for (int off = 1; off < 16; off <<= 1) rs += __shfl_xor(rs, off, 32);
            lrow[v] = lrow[v] * sc + rs;
            mrow[v] = mn;
#pragma unroll
            for (int dt = 0; dt < 4; ++dt) acc[dt][v] = acc[dt][v] * sc;
            const int pr = v + 8 * hlf;          // C-frag row
            Ps[(wid * 16 + pr) * 40 + l16]      = f2bf(p0);
            Ps[(wid * 16 + pr) * 40 + 16 + l16] = f2bf(p1);
        }
        __syncthreads();   // P visible before fragment reads

        // O += P (16x32) @ V (32x64)
        const v16bf ap = load_afrag(&Ps[(wid * 16 + l16) * 40], hlf);
#pragma unroll
        for (int dt = 0; dt < 4; ++dt)
            acc[dt] = wmma_bf16(ap,
                                load_bfrag(&Vbt[cur][(dt * 16 + l16) * 48], hlf),
                                acc[dt]);
    }

    // finalize: divide by l, store concat-head bf16 output
#pragma unroll
    for (int dt = 0; dt < 4; ++dt) {
#pragma unroll
        for (int v = 0; v < 8; ++v) {
            const int s = q0 + v + 8 * hlf;
            const float o = acc[dt][v] / lrow[v];
            outp[((size_t)(b * S_LEN + s)) * D_MODEL + h * D_K + dt * 16 + l16] = f2bf(o);
        }
    }
}

// ---------------------------------------------------------------------------
// Host-side orchestration
// ---------------------------------------------------------------------------
extern "C" void kernel_launch(void* const* d_in, const int* in_sizes, int n_in,
                              void* d_out, int out_size, void* d_ws, size_t ws_size,
                              hipStream_t stream) {
    (void)in_sizes; (void)n_in; (void)out_size; (void)ws_size;

    const float* x     = (const float*)d_in[0];
    const int*   mask  = (const int*)d_in[1];
    const float* Wq    = (const float*)d_in[2];
    const float* bq    = (const float*)d_in[3];
    const float* Wk    = (const float*)d_in[4];
    const float* bk    = (const float*)d_in[5];
    const float* Wv    = (const float*)d_in[6];
    const float* bv    = (const float*)d_in[7];
    const float* Wo    = (const float*)d_in[8];
    const float* bo    = (const float*)d_in[9];
    const float* W1    = (const float*)d_in[10];
    const float* b1    = (const float*)d_in[11];
    const float* W2    = (const float*)d_in[12];
    const float* b2    = (const float*)d_in[13];
    const float* ln1_g = (const float*)d_in[14];
    const float* ln1_b = (const float*)d_in[15];
    const float* ln2_g = (const float*)d_in[16];
    const float* ln2_b = (const float*)d_in[17];
    float* out = (float*)d_out;

    // workspace layout (bytes); region at +72MB holds [qk(32MB)|Vt(16MB)|attn(16MB)]
    // during attention, then the same 64MB is reused as the FFN hidden.
    char* ws = (char*)d_ws;
    const size_t MB = 1u << 20;
    bf16_t* WqT = (bf16_t*)(ws + 0 * MB);     // 2MB  [N][K]
    bf16_t* WkT = (bf16_t*)(ws + 2 * MB);     // 2MB
    bf16_t* WvT = (bf16_t*)(ws + 4 * MB);     // 2MB
    bf16_t* WoT = (bf16_t*)(ws + 6 * MB);     // 2MB
    bf16_t* W1T = (bf16_t*)(ws + 8 * MB);     // 8MB  [4096][1024]
    bf16_t* W2T = (bf16_t*)(ws + 16 * MB);    // 8MB  [1024][4096]
    bf16_t* h_b = (bf16_t*)(ws + 24 * MB);    // 16MB (ln1 out, later ln2 out)
    float*  x2  = (float*)(ws + 40 * MB);     // 32MB residual stream
    bf16_t* qk  = (bf16_t*)(ws + 72 * MB);    // 32MB [ROWS][2048] = Q|K
    bf16_t* Vt  = (bf16_t*)(ws + 104 * MB);   // 16MB [B][H][64][S]
    bf16_t* attn = (bf16_t*)(ws + 120 * MB);  // 16MB
    bf16_t* ffh = (bf16_t*)(ws + 72 * MB);    // 64MB (aliases qk/Vt/attn)

    // 1. transpose + convert weights to bf16 [N][K]
    cvt_transpose_kernel<<<dim3(32, 32),  256, 0, stream>>>(Wq, WqT, D_MODEL, D_MODEL);
    cvt_transpose_kernel<<<dim3(32, 32),  256, 0, stream>>>(Wk, WkT, D_MODEL, D_MODEL);
    cvt_transpose_kernel<<<dim3(32, 32),  256, 0, stream>>>(Wv, WvT, D_MODEL, D_MODEL);
    cvt_transpose_kernel<<<dim3(32, 32),  256, 0, stream>>>(Wo, WoT, D_MODEL, D_MODEL);
    cvt_transpose_kernel<<<dim3(128, 32), 256, 0, stream>>>(W1, W1T, D_MODEL, D_FF);
    cvt_transpose_kernel<<<dim3(32, 128), 256, 0, stream>>>(W2, W2T, D_FF, D_MODEL);

    // 2. h = LN1(x) -> bf16
    ln_bf16_kernel<<<ROWS, 256, 0, stream>>>(x, ln1_g, ln1_b, h_b);

    // 3. Q,K into fused [ROWS,2048] buffer; V written transposed
    dim3 g_proj(D_MODEL / 128, ROWS / 128);   // (8,64)
    gemm_bf16_kernel<<<g_proj, 256, 0, stream>>>(h_b, D_MODEL, WqT, D_MODEL, bq,
                                                 nullptr, 0, nullptr, qk + 0, nullptr,
                                                 2048, D_MODEL, 0);
    gemm_bf16_kernel<<<g_proj, 256, 0, stream>>>(h_b, D_MODEL, WkT, D_MODEL, bk,
                                                 nullptr, 0, nullptr, qk + 1024, nullptr,
                                                 2048, D_MODEL, 0);
    gemm_bf16_kernel<<<g_proj, 256, 0, stream>>>(h_b, D_MODEL, WvT, D_MODEL, bv,
                                                 nullptr, 0, nullptr, nullptr, Vt,
                                                 0, D_MODEL, 0);

    // 4. attention
    dim3 g_attn(S_LEN / 128, BATCH * N_HEADS);  // (16,64)
    attn_kernel<<<g_attn, 256, 0, stream>>>(qk, Vt, mask, attn);

    // 5. x2 = x + attn @ Wo + bo
    gemm_bf16_kernel<<<g_proj, 256, 0, stream>>>(attn, D_MODEL, WoT, D_MODEL, bo,
                                                 x, D_MODEL, x2, nullptr, nullptr,
                                                 D_MODEL, D_MODEL, 0);

    // 6. h2 = LN2(x2) -> bf16 (reuse h_b)
    ln_bf16_kernel<<<ROWS, 256, 0, stream>>>(x2, ln2_g, ln2_b, h_b);

    // 7. ffh = GELU(h2 @ W1 + b1) -> bf16
    dim3 g_ff1(D_FF / 128, ROWS / 128);   // (32,64)
    gemm_bf16_kernel<<<g_ff1, 256, 0, stream>>>(h_b, D_MODEL, W1T, D_MODEL, b1,
                                                nullptr, 0, nullptr, ffh, nullptr,
                                                D_FF, D_MODEL, 1);

    // 8. out = x2 + ffh @ W2 + b2
    gemm_bf16_kernel<<<g_proj, 256, 0, stream>>>(ffh, D_FF, W2T, D_FF, b2,
                                                 x2, D_MODEL, out, nullptr, nullptr,
                                                 D_MODEL, D_FF, 0);
}
